// GNN_25769804267
// MI455X (gfx1250) — compile-verified, matrix operand order checked by reading, not run on hardware
//
#include <hip/hip_runtime.h>
#include <cstdint>
#include <cstddef>

#define NNODES 100000
#define NEDGES 1600000
#define HID 64

// ---------------- WMMA types ----------------
typedef __attribute__((ext_vector_type(16))) __bf16 v16bf;
typedef __attribute__((ext_vector_type(8)))  float  v8f;

union ABReg { unsigned int u[8]; v16bf v; };

__device__ __forceinline__ unsigned short f2bf(float f) {
    union { float f; unsigned int u; } c; c.f = f;
    unsigned int r = c.u + 0x7FFFu + ((c.u >> 16) & 1u);   // round-to-nearest-even
    return (unsigned short)(r >> 16);
}
__device__ __forceinline__ unsigned int pack2bf(float a, float b) {
    return (unsigned int)f2bf(a) | ((unsigned int)f2bf(b) << 16);
}
__device__ __forceinline__ float silu_f(float x) {
    return x / (1.0f + __expf(-x));
}

// Load a 16x32 bf16 A-operand K-chunk for this lane from a bf16 row pointer.
// ISA 16-bit A layout: lane(0-15)=row M, khalf=lane>>4 selects K 0-7/8-15 (V0-3)
// and 16-23/24-31 (V4-7); K pairs are contiguous => two b128 loads.
__device__ __forceinline__ void load_a_row(ABReg& a, const unsigned short* p /*row base + chunk*32 + khalf*8*/) {
    uint4 x = *(const uint4*)p;
    uint4 y = *(const uint4*)(p + 16);
    a.u[0]=x.x; a.u[1]=x.y; a.u[2]=x.z; a.u[3]=x.w;
    a.u[4]=y.x; a.u[5]=y.y; a.u[6]=y.z; a.u[7]=y.w;
}
// Load a pre-packed B-operand block (1KB = 32 lanes x 8 dwords, lane-major).
__device__ __forceinline__ void load_b_pk(ABReg& b, const unsigned int* base, int blk, int lane) {
    const uint4* q = (const uint4*)(base + (size_t)(blk * 32 + lane) * 8);
    uint4 x = q[0], y = q[1];
    b.u[0]=x.x; b.u[1]=x.y; b.u[2]=x.z; b.u[3]=x.w;
    b.u[4]=y.x; b.u[5]=y.y; b.u[6]=y.z; b.u[7]=y.w;
}
__device__ __forceinline__ v8f wmma_bf16(const ABReg& a, const ABReg& b, v8f c) {
    return __builtin_amdgcn_wmma_f32_16x16x32_bf16(false, a.v, false, b.v, (short)0, c, false, false);
}

// ---------------- weight packing ----------------
// Packs row-major fp32 W[Kdim x Ncols] into bf16 WMMA-B lane layout, zero-padded.
// dst dword index = ((c*Ntiles + t)*32 + l)*8 + d
__global__ void gnn_pack_b(const float* __restrict__ W, unsigned int* __restrict__ dst,
                           int Kdim, int Ncols, int Ntiles, int Kchunks) {
    int tid = blockIdx.x * blockDim.x + threadIdx.x;
    int total = Kchunks * Ntiles * 256;
    if (tid >= total) return;
    int d  = tid & 7;
    int l  = (tid >> 3) & 31;
    int ct = tid >> 8;
    int t  = ct % Ntiles;
    int c  = ct / Ntiles;
    int n      = t * 16 + (l & 15);
    int khalf  = l >> 4;
    int klocal = ((d & 4) ? 16 : 0) + khalf * 8 + (d & 3) * 2;
    int k      = c * 32 + klocal;
    float v0 = (k     < Kdim && n < Ncols) ? W[(size_t)k * Ncols + n]       : 0.0f;
    float v1 = (k + 1 < Kdim && n < Ncols) ? W[(size_t)(k + 1) * Ncols + n] : 0.0f;
    dst[tid] = pack2bf(v0, v1);
}

// ---------------- small elementwise kernels ----------------
__global__ void gnn_embed(const float* __restrict__ loc, const float* __restrict__ vel,
                          const float* __restrict__ ew, const float* __restrict__ eb,
                          float* __restrict__ h, unsigned short* __restrict__ hb) {
    int tid = blockIdx.x * blockDim.x + threadIdx.x;
    if (tid >= NNODES * HID) return;
    int node = tid >> 6, f = tid & 63;
    float x0 = loc[node*3+0], x1 = loc[node*3+1], x2 = loc[node*3+2];
    float x3 = vel[node*3+0], x4 = vel[node*3+1], x5 = vel[node*3+2];
    float s = eb[f];
    s += x0*ew[0*HID+f] + x1*ew[1*HID+f] + x2*ew[2*HID+f]
       + x3*ew[3*HID+f] + x4*ew[4*HID+f] + x5*ew[5*HID+f];
    h[tid]  = s;
    hb[tid] = f2bf(s);
}

__global__ void gnn_zero(float* __restrict__ p, int n) {
    int tid = blockIdx.x * blockDim.x + threadIdx.x;
    if (tid < n) p[tid] = 0.0f;
}

__global__ void gnn_cvt_bf16(const float* __restrict__ src, unsigned short* __restrict__ dst, int n) {
    int tid = blockIdx.x * blockDim.x + threadIdx.x;
    if (tid < n) dst[tid] = f2bf(src[tid]);
}

// ---------------- edge MLP + scatter (1 wave = 16 edges) ----------------
__global__ __launch_bounds__(128) void gnn_edge(
    const unsigned short* __restrict__ hb, const float* __restrict__ eattr,
    const int* __restrict__ eidx,               // [2*E], rows then cols
    const unsigned int* __restrict__ pw1, const float* __restrict__ b1,  // 130->64
    const unsigned int* __restrict__ pw2, const float* __restrict__ b2,  // 64->64
    float* __restrict__ agg, int E) {
    __shared__ unsigned short smem[4][16 * HID];   // 2KB per wave bounce buffer
    const int lane  = threadIdx.x & 31;
    const int w     = threadIdx.x >> 5;
    const int tile  = blockIdx.x * 4 + w;
    const int ebase = tile * 16;
    if (ebase >= E) return;
    const int mrow  = lane & 15;
    const int khalf = lane >> 4;
    const int e_m   = ebase + mrow;
    const int rsrc  = eidx[e_m];
    const int cdst  = eidx[E + e_m];
    const unsigned short* pr = hb + (size_t)rsrc * HID;
    const unsigned short* pc = hb + (size_t)cdst * HID;

    // ---- layer 1: [16 x 130] @ [130 x 64], K chunks: h[row](2) h[col](2) attr(1)
    v8f acc[4];
#pragma unroll
    for (int t = 0; t < 4; ++t) {
        float bv = b1[t * 16 + mrow];
        v8f ai = {bv, bv, bv, bv, bv, bv, bv, bv};
        acc[t] = ai;
    }
    ABReg a, b;
#pragma unroll
    for (int kc = 0; kc < 5; ++kc) {
        if (kc == 4) {
            unsigned int u0 = 0;
            if (khalf == 0) u0 = pack2bf(eattr[(size_t)e_m * 2], eattr[(size_t)e_m * 2 + 1]);
            a.u[0] = u0;
#pragma unroll
            for (int j = 1; j < 8; ++j) a.u[j] = 0;
        } else {
            const unsigned short* p = (kc < 2 ? pr : pc) + (kc & 1) * 32 + khalf * 8;
            load_a_row(a, p);
        }
#pragma unroll
        for (int t = 0; t < 4; ++t) {
            load_b_pk(b, pw1, kc * 4 + t, lane);
            acc[t] = wmma_bf16(a, b, acc[t]);
        }
    }
    // SiLU, bounce D(C-layout) -> LDS(bf16 row-major) for A-layout reload
#pragma unroll
    for (int t = 0; t < 4; ++t)
#pragma unroll
        for (int j = 0; j < 8; ++j)
            smem[w][(j + khalf * 8) * HID + t * 16 + mrow] = f2bf(silu_f(acc[t][j]));
    __builtin_amdgcn_wave_barrier();   // per-wave LDS ops are in-order; just fence scheduling

    // ---- layer 2: [16 x 64] @ [64 x 64]
    v8f acc2[4];
#pragma unroll
    for (int t = 0; t < 4; ++t) {
        float bv = b2[t * 16 + mrow];
        v8f ai = {bv, bv, bv, bv, bv, bv, bv, bv};
        acc2[t] = ai;
    }
#pragma unroll
    for (int kc = 0; kc < 2; ++kc) {
        load_a_row(a, &smem[w][mrow * HID + kc * 32 + khalf * 8]);
#pragma unroll
        for (int t = 0; t < 4; ++t) {
            load_b_pk(b, pw2, kc * 4 + t, lane);
            acc2[t] = wmma_bf16(a, b, acc2[t]);
        }
    }
    // SiLU + scatter-add to agg[row]
    int rr[8];
#pragma unroll
    for (int j = 0; j < 8; ++j) rr[j] = eidx[ebase + khalf * 8 + j];
#pragma unroll
    for (int t = 0; t < 4; ++t)
#pragma unroll
        for (int j = 0; j < 8; ++j) {
            float x = silu_f(acc2[t][j]);
            unsafeAtomicAdd(&agg[(size_t)rr[j] * HID + t * 16 + mrow], x);
        }
}

// ---------------- node MLP + residual (1 wave = 16 nodes) ----------------
__global__ __launch_bounds__(128) void gnn_node(
    unsigned short* __restrict__ hb, float* __restrict__ h,
    const unsigned short* __restrict__ aggb,
    const unsigned int* __restrict__ pw1, const float* __restrict__ b1,  // 128->64
    const unsigned int* __restrict__ pw2, const float* __restrict__ b2,  // 64->64
    int ntiles) {
    __shared__ unsigned short smem[4][16 * HID];
    const int lane  = threadIdx.x & 31;
    const int w     = threadIdx.x >> 5;
    const int tile  = blockIdx.x * 4 + w;
    if (tile >= ntiles) return;
    const int nbase = tile * 16;
    const int mrow  = lane & 15;
    const int khalf = lane >> 4;
    const unsigned short* pr = hb   + (size_t)(nbase + mrow) * HID;
    const unsigned short* pa = aggb + (size_t)(nbase + mrow) * HID;

    v8f acc[4];
#pragma unroll
    for (int t = 0; t < 4; ++t) {
        float bv = b1[t * 16 + mrow];
        v8f ai = {bv, bv, bv, bv, bv, bv, bv, bv};
        acc[t] = ai;
    }
    ABReg a, b;
#pragma unroll
    for (int kc = 0; kc < 4; ++kc) {
        const unsigned short* p = (kc < 2 ? pr + kc * 32 : pa + (kc - 2) * 32) + khalf * 8;
        load_a_row(a, p);
#pragma unroll
        for (int t = 0; t < 4; ++t) {
            load_b_pk(b, pw1, kc * 4 + t, lane);
            acc[t] = wmma_bf16(a, b, acc[t]);
        }
    }
#pragma unroll
    for (int t = 0; t < 4; ++t)
#pragma unroll
        for (int j = 0; j < 8; ++j)
            smem[w][(j + khalf * 8) * HID + t * 16 + mrow] = f2bf(silu_f(acc[t][j]));
    __builtin_amdgcn_wave_barrier();

    v8f acc2[4];
#pragma unroll
    for (int t = 0; t < 4; ++t) {
        float bv = b2[t * 16 + mrow];
        v8f ai = {bv, bv, bv, bv, bv, bv, bv, bv};
        acc2[t] = ai;
    }
#pragma unroll
    for (int kc = 0; kc < 2; ++kc) {
        load_a_row(a, &smem[w][mrow * HID + kc * 32 + khalf * 8]);
#pragma unroll
        for (int t = 0; t < 4; ++t) {
            load_b_pk(b, pw2, kc * 4 + t, lane);
            acc2[t] = wmma_bf16(a, b, acc2[t]);
        }
    }
    // residual: h += out   (no activation on second node layer)
#pragma unroll
    for (int t = 0; t < 4; ++t)
#pragma unroll
        for (int j = 0; j < 8; ++j) {
            int mm = j + khalf * 8;
            int nn = t * 16 + mrow;
            size_t idx = (size_t)(nbase + mm) * HID + nn;
            float hv = h[idx] + acc2[t][j];
            h[idx]  = hv;
            hb[idx] = f2bf(hv);
        }
}

// ---------------- decoder (1 wave = 16 nodes) ----------------
__global__ __launch_bounds__(128) void gnn_decode(
    const unsigned short* __restrict__ hb,
    const unsigned int* __restrict__ pw1, const float* __restrict__ b1,  // 64->64
    const unsigned int* __restrict__ pw2, const float* __restrict__ b2,  // 64->3 (padded)
    float* __restrict__ out, int ntiles) {
    __shared__ unsigned short smem[4][16 * HID];
    const int lane  = threadIdx.x & 31;
    const int w     = threadIdx.x >> 5;
    const int tile  = blockIdx.x * 4 + w;
    if (tile >= ntiles) return;
    const int nbase = tile * 16;
    const int mrow  = lane & 15;
    const int khalf = lane >> 4;
    const unsigned short* pr = hb + (size_t)(nbase + mrow) * HID;

    v8f acc[4];
#pragma unroll
    for (int t = 0; t < 4; ++t) {
        float bv = b1[t * 16 + mrow];
        v8f ai = {bv, bv, bv, bv, bv, bv, bv, bv};
        acc[t] = ai;
    }
    ABReg a, b;
#pragma unroll
    for (int kc = 0; kc < 2; ++kc) {
        load_a_row(a, pr + kc * 32 + khalf * 8);
#pragma unroll
        for (int t = 0; t < 4; ++t) {
            load_b_pk(b, pw1, kc * 4 + t, lane);
            acc[t] = wmma_bf16(a, b, acc[t]);
        }
    }
#pragma unroll
    for (int t = 0; t < 4; ++t)
#pragma unroll
        for (int j = 0; j < 8; ++j)
            smem[w][(j + khalf * 8) * HID + t * 16 + mrow] = f2bf(silu_f(acc[t][j]));
    __builtin_amdgcn_wave_barrier();

    float bv = (mrow < 3) ? b2[mrow] : 0.0f;
    v8f accd = {bv, bv, bv, bv, bv, bv, bv, bv};
#pragma unroll
    for (int kc = 0; kc < 2; ++kc) {
        load_a_row(a, &smem[w][mrow * HID + kc * 32 + khalf * 8]);
        load_b_pk(b, pw2, kc, lane);      // Ntiles==1 packing
        accd = wmma_bf16(a, b, accd);
    }
    if (mrow < 3) {
#pragma unroll
        for (int j = 0; j < 8; ++j) {
            int mm = j + khalf * 8;
            out[(size_t)(nbase + mm) * 3 + mrow] = accd[j];
        }
    }
}

// ---------------- host launcher ----------------
extern "C" void kernel_launch(void* const* d_in, const int* in_sizes, int n_in,
                              void* d_out, int out_size, void* d_ws, size_t ws_size,
                              hipStream_t stream) {
    (void)in_sizes; (void)n_in; (void)out_size; (void)ws_size;
    const float* loc   = (const float*)d_in[1];
    const float* vel   = (const float*)d_in[2];
    const float* eattr = (const float*)d_in[3];
    const int*   edges = (const int*)d_in[4];
    const float* emb_w = (const float*)d_in[5];
    const float* emb_b = (const float*)d_in[6];
    const float* ew1 = (const float*)d_in[7];
    const float* eb1 = (const float*)d_in[8];
    const float* ew2 = (const float*)d_in[9];
    const float* eb2 = (const float*)d_in[10];
    const float* nw1 = (const float*)d_in[11];
    const float* nb1 = (const float*)d_in[12];
    const float* nw2 = (const float*)d_in[13];
    const float* nb2 = (const float*)d_in[14];
    const float* dw1 = (const float*)d_in[15];
    const float* db1 = (const float*)d_in[16];
    const float* dw2 = (const float*)d_in[17];
    const float* db2 = (const float*)d_in[18];
    float* out = (float*)d_out;

    char* ws = (char*)d_ws; size_t o = 0;
    auto take = [&](size_t bytes) -> void* {
        o = (o + 255) & ~(size_t)255;
        void* p = ws + o; o += bytes; return p;
    };
    float*          h    = (float*)take((size_t)NNODES * HID * 4);
    unsigned short* hb   = (unsigned short*)take((size_t)NNODES * HID * 2);
    float*          agg  = (float*)take((size_t)NNODES * HID * 4);
    unsigned short* aggb = (unsigned short*)take((size_t)NNODES * HID * 2);
    unsigned int* pw_ew1 = (unsigned int*)take((size_t)4 * 5120 * 4);  // 5 chunks x 4 tiles
    unsigned int* pw_ew2 = (unsigned int*)take((size_t)4 * 2048 * 4);  // 2 x 4
    unsigned int* pw_nw1 = (unsigned int*)take((size_t)4 * 4096 * 4);  // 4 x 4
    unsigned int* pw_nw2 = (unsigned int*)take((size_t)4 * 2048 * 4);  // 2 x 4
    unsigned int* pw_d1  = (unsigned int*)take((size_t)2048 * 4);      // 2 x 4
    unsigned int* pw_d2  = (unsigned int*)take((size_t)512 * 4);       // 2 x 1

    auto pack = [&](const float* W, unsigned int* dst, int Kdim, int Ncols, int Nt, int Kc) {
        int total = Kc * Nt * 256;
        gnn_pack_b<<<(total + 255) / 256, 256, 0, stream>>>(W, dst, Kdim, Ncols, Nt, Kc);
    };
    for (int i = 0; i < 4; ++i) {
        pack(ew1 + (size_t)i * 130 * 64, pw_ew1 + (size_t)i * 5120, 130, 64, 4, 5);
        pack(ew2 + (size_t)i * 64 * 64,  pw_ew2 + (size_t)i * 2048, 64, 64, 4, 2);
        pack(nw1 + (size_t)i * 128 * 64, pw_nw1 + (size_t)i * 4096, 128, 64, 4, 4);
        pack(nw2 + (size_t)i * 64 * 64,  pw_nw2 + (size_t)i * 2048, 64, 64, 4, 2);
    }
    pack(dw1, pw_d1, 64, 64, 4, 2);
    pack(dw2, pw_d2, 64, 3, 1, 2);

    const int NH = NNODES * HID;
    gnn_embed<<<(NH + 255) / 256, 256, 0, stream>>>(loc, vel, emb_w, emb_b, h, hb);

    const int etiles = NEDGES / 16;            // 100000
    const int ntiles = NNODES / 16;            // 6250
    for (int i = 0; i < 4; ++i) {
        gnn_zero<<<(NH + 255) / 256, 256, 0, stream>>>(agg, NH);
        gnn_edge<<<(etiles + 3) / 4, 128, 0, stream>>>(
            hb, eattr, edges,
            pw_ew1 + (size_t)i * 5120, eb1 + i * 64,
            pw_ew2 + (size_t)i * 2048, eb2 + i * 64,
            agg, NEDGES);
        gnn_cvt_bf16<<<(NH + 255) / 256, 256, 0, stream>>>(agg, aggb, NH);
        gnn_node<<<(ntiles + 3) / 4, 128, 0, stream>>>(
            hb, h, aggb,
            pw_nw1 + (size_t)i * 4096, nb1 + i * 64,
            pw_nw2 + (size_t)i * 2048, nb2 + i * 64,
            ntiles);
    }
    gnn_decode<<<(ntiles + 3) / 4, 128, 0, stream>>>(hb, pw_d1, db1, pw_d2, db2, out, ntiles);
}
